// EdgeBipartiteDenoiserV4_HF_Lite_19000935317554
// MI455X (gfx1250) — matrix operand order repaired and checked by reading, compiled, not run on hardware
//
#include <hip/hip_runtime.h>
#include <hip/hip_bf16.h>
#include <math.h>

// ---------------------------------------------------------------------------
// Problem constants (from reference)
// ---------------------------------------------------------------------------
#define B_   256
#define K_   128
#define H_   192
#define NH_  4
#define HD_  48          // H_/NH_
#define MID_ 48
#define FFN_ 384
#define N_   (B_ * K_)   // 32768
#define LN_EPS 1e-5f

typedef __attribute__((ext_vector_type(16))) __bf16 v16bf;
typedef __attribute__((ext_vector_type(8)))  float  v8f;

// ---------------------------------------------------------------------------
// CDNA5 async copy global->LDS (ASYNCcnt-tracked), inline asm for portability.
// LDS generic pointers carry the LDS byte offset in the low 32 bits.
// ---------------------------------------------------------------------------
__device__ __forceinline__ void async_copy_b128(const void* gptr, void* lptr) {
    asm volatile("global_load_async_to_lds_b128 %0, %1, off"
                 :: "v"((unsigned int)(uintptr_t)lptr),
                    "v"((unsigned long long)(uintptr_t)gptr)
                 : "memory");
}

template<int N>
__device__ __forceinline__ void async_wait() {
    asm volatile("s_wait_asynccnt %0" :: "n"(N) : "memory");
}

// ---------------------------------------------------------------------------
// WMMA helpers: bf16 16x16x32, f32 accumulate
// ---------------------------------------------------------------------------
__device__ __forceinline__ v8f wmma_bf16(v16bf a, v16bf b, v8f c) {
    return __builtin_amdgcn_wmma_f32_16x16x32_bf16(
        /*neg_a=*/false, a, /*neg_b=*/false, b,
        /*c_mod=*/(short)0, c, /*reuse_a=*/false, /*reuse_b=*/false);
}

// A fragment: 16x32 bf16, rows contiguous in K in LDS (row stride srow bf16s).
__device__ __forceinline__ v16bf load_frag_a(const __bf16* base, int rowBase,
                                             int srow, int koff, int lane) {
    const int row = rowBase + (lane & 15);
    const int kh  = (lane >> 4) & 1;
    const __bf16* p = base + (size_t)row * srow + koff + kh * 8;
    union { v16bf v; uint4 u[2]; } r;
    r.u[0] = *reinterpret_cast<const uint4*>(p);
    r.u[1] = *reinterpret_cast<const uint4*>(p + 16);
    return r.v;
}

// B fragment: 32x16 bf16 from LDS stored [n][k] (row stride srow bf16s).
__device__ __forceinline__ v16bf load_frag_b(const __bf16* base, int colBase,
                                             int srow, int koff, int lane) {
    const int col = colBase + (lane & 15);
    const int kh  = (lane >> 4) & 1;
    const __bf16* p = base + (size_t)col * srow + koff + kh * 8;
    union { v16bf v; uint4 u[2]; } r;
    r.u[0] = *reinterpret_cast<const uint4*>(p);
    r.u[1] = *reinterpret_cast<const uint4*>(p + 16);
    return r.v;
}

// ---------------------------------------------------------------------------
// One-time conversion kernels (tiny vs. main GEMMs)
// ---------------------------------------------------------------------------
__global__ __launch_bounds__(256)
void f32_to_bf16_kernel(const float* __restrict__ in, __bf16* __restrict__ out,
                        int nQuads) {
    for (int i = blockIdx.x * 256 + threadIdx.x; i < nQuads;
         i += gridDim.x * 256) {
        const float4 f = reinterpret_cast<const float4*>(in)[i];
        __bf16 t[4] = {(__bf16)f.x, (__bf16)f.y, (__bf16)f.z, (__bf16)f.w};
        reinterpret_cast<uint2*>(out)[i] = *reinterpret_cast<const uint2*>(t);
    }
}

// Wt[n][k] = (bf16) W[k][n]   (weights pre-transposed for straight B staging)
__global__ __launch_bounds__(256)
void transpose_cvt_kernel(const float* __restrict__ W, __bf16* __restrict__ Wt,
                          int Kd, int Nd) {
    const int total = Kd * Nd;
    for (int i = blockIdx.x * 256 + threadIdx.x; i < total;
         i += gridDim.x * 256) {
        const int n = i / Kd;
        const int k = i - n * Kd;
        Wt[i] = (__bf16)W[(size_t)k * Nd + n];
    }
}

// ---------------------------------------------------------------------------
// Tiled GEMM:  C[M x Nd] = A[M x Kd] * Bt[Nd x Kd]^T   (A,Bt bf16)
// 128 threads (4 waves). 128x64 tile, KC=64. Each wave owns a 32x64 strip
// (2x4 wmma tiles, 16 WMMAs per stage between barriers). A/B tiles streamed
// with async-to-LDS, double buffered. Requires M%128==0, Nd%64==0, Kd%64==0.
// ---------------------------------------------------------------------------
template<bool HAS_BIAS, bool DO_SILU, bool OUT_F32, bool OUT_BF16>
__global__ __launch_bounds__(128)
void gemm_bf16_kernel(const __bf16* __restrict__ A, const __bf16* __restrict__ Bt,
                      const float* __restrict__ bias,
                      float* __restrict__ C, __bf16* __restrict__ Cb,
                      int Kd, int Nd) {
    constexpr int BM = 128, BN = 64, KC = 64;
    __shared__ __bf16 As[2][BM][KC];   // 32 KB
    __shared__ __bf16 Bs[2][BN][KC];   // 16 KB

    const int tid  = threadIdx.x;
    const int lane = tid & 31;
    const int wave = tid >> 5;
    const int wm   = wave * 32;                 // wave's row strip
    const int m0   = blockIdx.x * BM;
    const int n0   = blockIdx.y * BN;
    const int nIter = Kd / KC;

    // Stage one (A,B) tile pair into buffer `buf` via async copies.
    // (128+64) rows x 8 chunks = 1536 16B chunks / 128 thr = 12 per thread,
    // uniform A/B split per iteration.
    auto stage = [&](int buf, int k0) {
        #pragma unroll
        for (int i = tid; i < 1536; i += 128) {
            const int isB = (i >= 1024) ? 1 : 0;
            const int j   = isB ? i - 1024 : i;
            const int row = j >> 3;
            const int seg = (j & 7) << 3;          // bf16 elements
            const __bf16* g = isB
                ? Bt + (size_t)(n0 + row) * Kd + k0 + seg
                : A  + (size_t)(m0 + row) * Kd + k0 + seg;
            __bf16* l = isB ? &Bs[buf][row][seg] : &As[buf][row][seg];
            async_copy_b128(g, l);
        }
    };

    v8f acc[2][4] = {};
    stage(0, 0);

    for (int it = 0; it < nIter; ++it) {
        const int buf = it & 1;
        if (it + 1 < nIter) {
            stage(buf ^ 1, (it + 1) * KC);
            async_wait<12>();    // 12 outstanding = next stage; current done
        } else {
            async_wait<0>();
        }
        __syncthreads();

        #pragma unroll
        for (int kk = 0; kk < KC; kk += 32) {
            const v16bf a0 = load_frag_a(&As[buf][0][0], wm,      KC, kk, lane);
            const v16bf a1 = load_frag_a(&As[buf][0][0], wm + 16, KC, kk, lane);
            #pragma unroll
            for (int nt = 0; nt < 4; ++nt) {
                const v16bf b = load_frag_b(&Bs[buf][0][0], nt * 16, KC, kk, lane);
                acc[0][nt] = wmma_bf16(a0, b, acc[0][nt]);
                acc[1][nt] = wmma_bf16(a1, b, acc[1][nt]);
            }
        }
        __syncthreads();
    }

    // Epilogue: C layout -> lane(L&15)=n, VGPR r -> m = r + 8*(L>>4).
    #pragma unroll
    for (int mt = 0; mt < 2; ++mt)
        #pragma unroll
        for (int nt = 0; nt < 4; ++nt)
            #pragma unroll
            for (int r = 0; r < 8; ++r) {
                const int row = m0 + wm + mt * 16 + ((lane >> 4) << 3) + r;
                const int col = n0 + nt * 16 + (lane & 15);
                float v = acc[mt][nt][r];
                if (HAS_BIAS) v += bias[col];
                if (DO_SILU)  v = v / (1.0f + __expf(-v));
                if (OUT_F32)  C[(size_t)row * Nd + col] = v;
                if (OUT_BF16) Cb[(size_t)row * Nd + col] = (__bf16)v;
            }
}

// ---------------------------------------------------------------------------
// Pairwise-feature MLP bias -> bf16 bias[b][h][m][n], one block per batch.
// ---------------------------------------------------------------------------
__global__ __launch_bounds__(256)
void pair_bias_kernel(const float* __restrict__ cap,
                      const float* __restrict__ fixedr,
                      const float* __restrict__ unitr,
                      const float* __restrict__ tierr,
                      const float* __restrict__ tierraw,
                      const float* __restrict__ pW1, const float* __restrict__ pb1,
                      const float* __restrict__ pW2, const float* __restrict__ pb2,
                      __bf16* __restrict__ biasOut) {
    const int b   = blockIdx.x;
    const int tid = threadIdx.x;

    __shared__ float s_cap[K_], s_fix[K_], s_unit[K_], s_trel[K_], s_traw[K_];
    __shared__ float w1[5][MID_], b1v[MID_], w2[MID_][4], b2v[4];
    __shared__ float s_capmax;

    if (tid < K_) {
        const int i = b * K_ + tid;
        s_cap[tid]  = cap[i];
        s_fix[tid]  = fixedr[i];
        s_unit[tid] = unitr[i];
        s_trel[tid] = tierr[i];
        s_traw[tid] = tierraw[i];
    }
    for (int i = tid; i < 5 * MID_; i += 256) w1[i / MID_][i % MID_] = pW1[i];
    for (int i = tid; i < MID_; i += 256)     b1v[i] = pb1[i];
    for (int i = tid; i < MID_ * 4; i += 256) w2[i >> 2][i & 3] = pW2[i];
    if (tid < 4) b2v[tid] = pb2[tid];
    __syncthreads();

    if (tid == 0) {
        float m = 0.0f;
        for (int i = 0; i < K_; ++i) m = fmaxf(m, s_cap[i]);
        s_capmax = fmaxf(m, 1e-6f);
    }
    __syncthreads();
    const float invmax = 1.0f / s_capmax;

    const size_t base = (size_t)b * NH_ * K_ * K_;
    for (int p = tid; p < K_ * K_; p += 256) {
        const int m = p >> 7;
        const int n = p & 127;
        const float f0 = (s_traw[m] == s_traw[n]) ? 1.0f : 0.0f;
        const float f1 = fabsf((s_cap[m] - s_cap[n]) * invmax);
        const float f2 = fabsf(s_fix[m]  - s_fix[n]);
        const float f3 = fabsf(s_unit[m] - s_unit[n]);
        const float f4 = fabsf(s_trel[m] - s_trel[n]);
        float o0 = b2v[0], o1 = b2v[1], o2 = b2v[2], o3 = b2v[3];
        #pragma unroll 4
        for (int j = 0; j < MID_; ++j) {
            float mj = b1v[j] + f0 * w1[0][j] + f1 * w1[1][j] + f2 * w1[2][j]
                              + f3 * w1[3][j] + f4 * w1[4][j];
            mj = mj / (1.0f + __expf(-mj));      // silu
            o0 += mj * w2[j][0];
            o1 += mj * w2[j][1];
            o2 += mj * w2[j][2];
            o3 += mj * w2[j][3];
        }
        const size_t off = (size_t)m * K_ + n;
        biasOut[base + 0 * (K_ * K_) + off] = (__bf16)o0;
        biasOut[base + 1 * (K_ * K_) + off] = (__bf16)o1;
        biasOut[base + 2 * (K_ * K_) + off] = (__bf16)o2;
        biasOut[base + 3 * (K_ * K_) + off] = (__bf16)o3;
    }
}

// ---------------------------------------------------------------------------
// Fused attention: one block per (batch, head). 256 threads = 8 waves.
// qkv is bf16 [row][576]; Q/K staged via async-to-LDS (d padded 48->64),
// V transposed into LDS; P reuses the Q/K LDS region.
// ---------------------------------------------------------------------------
__global__ __launch_bounds__(256)
void attn_kernel(const __bf16* __restrict__ qkvB,
                 const __bf16* __restrict__ biasB,
                 __bf16* __restrict__ attnOutB) {
    const int b = blockIdx.x >> 2;
    const int h = blockIdx.x & 3;

    __shared__ __bf16 sQK[2][K_][64];   // 32 KB; reused as P[128][128] later
    __shared__ __bf16 sVt[64][K_];      // 16 KB; Vt[d][key]
    __bf16 (*Ps)[K_] = reinterpret_cast<__bf16 (*)[K_]>(&sQK[0][0][0]);

    const int tid  = threadIdx.x;
    const int lane = tid & 31;
    const int wave = tid >> 5;

    const __bf16* gbase = qkvB + (size_t)b * K_ * (3 * H_) + (size_t)h * HD_;

    // ---- async stage Q and K rows (48 bf16 = 6 x 16B chunks per row) ----
    #pragma unroll
    for (int i = tid; i < 2 * K_ * 6; i += 256) {
        const int m   = (i >= K_ * 6) ? 1 : 0;          // uniform per step
        const int j   = i - m * (K_ * 6);
        const int row = j / 6;
        const int seg = (j - row * 6) << 3;             // bf16 elements
        const __bf16* g = gbase + (size_t)row * (3 * H_) + m * H_ + seg;
        async_copy_b128(g, &sQK[m][row][seg]);
    }
    // ---- zero-pad d = 48..63 of Q and K (two 16B segs per row) ----
    #pragma unroll
    for (int i = tid; i < 512; i += 256) {
        const int m    = i >> 8;                        // uniform per step
        const int j    = i & 255;
        const int row  = j >> 1;
        const int hseg = (j & 1) << 3;
        const uint4 z = {0u, 0u, 0u, 0u};
        *reinterpret_cast<uint4*>(&sQK[m][row][48 + hseg]) = z;
    }
    // ---- V transpose: sVt[d][key] = V[key][d] ----
    const __bf16* gv = gbase + 2 * H_;
    for (int i = tid; i < K_ * HD_; i += 256) {
        const int row = i / HD_;
        const int c   = i - row * HD_;
        sVt[c][row] = gv[(size_t)row * (3 * H_) + c];
    }
    async_wait<0>();
    __syncthreads();

    // ---- scores: wave handles query rows [16*wave, +16), 8 key tiles ----
    const int mrow = wave * 16;
    v8f acc[8] = {};
    #pragma unroll
    for (int d0 = 0; d0 < 64; d0 += 32) {
        const v16bf a = load_frag_a(&sQK[0][0][0], mrow, 64, d0, lane);
        #pragma unroll
        for (int t = 0; t < 8; ++t) {
            const v16bf bf = load_frag_b(&sQK[1][0][0], t * 16, 64, d0, lane);
            acc[t] = wmma_bf16(a, bf, acc[t]);
        }
    }

    // ---- scale + pair bias ----
    const float scale = 0.14433756729740643f;   // 1/sqrt(48)
    const __bf16* bb = biasB + ((size_t)(b * NH_ + h)) * (K_ * K_);
    #pragma unroll
    for (int t = 0; t < 8; ++t)
        #pragma unroll
        for (int r = 0; r < 8; ++r) {
            const int row = mrow + ((lane >> 4) << 3) + r;
            const int col = t * 16 + (lane & 15);
            acc[t][r] = acc[t][r] * scale + (float)bb[(size_t)row * K_ + col];
        }

    // ---- softmax over 128 cols (row lives in 16 lanes of a half-wave) ----
    #pragma unroll
    for (int r = 0; r < 8; ++r) {
        float mx = -3.4e38f;
        #pragma unroll
        for (int t = 0; t < 8; ++t) mx = fmaxf(mx, acc[t][r]);
        #pragma unroll
        for (int off = 8; off >= 1; off >>= 1)
            mx = fmaxf(mx, __shfl_xor(mx, off, 32));
        float sum = 0.0f;
        #pragma unroll
        for (int t = 0; t < 8; ++t) {
            const float e = __expf(acc[t][r] - mx);
            acc[t][r] = e;
            sum += e;
        }
        #pragma unroll
        for (int off = 8; off >= 1; off >>= 1)
            sum += __shfl_xor(sum, off, 32);
        const float inv = 1.0f / sum;
        #pragma unroll
        for (int t = 0; t < 8; ++t) acc[t][r] *= inv;
    }

    __syncthreads();   // all waves done reading Q/K before P overwrites it

    #pragma unroll
    for (int t = 0; t < 8; ++t)
        #pragma unroll
        for (int r = 0; r < 8; ++r) {
            const int row = mrow + ((lane >> 4) << 3) + r;
            const int col = t * 16 + (lane & 15);
            Ps[row][col] = (__bf16)acc[t][r];
        }
    __syncthreads();

    // ---- out = P @ V  (128x128 @ 128x48, 3 col tiles) ----
    v8f o[3] = {};
    #pragma unroll
    for (int k0 = 0; k0 < K_; k0 += 32) {
        const v16bf a = load_frag_a(&Ps[0][0], mrow, K_, k0, lane);
        #pragma unroll
        for (int n = 0; n < 3; ++n) {
            const v16bf bf = load_frag_b(&sVt[0][0], n * 16, K_, k0, lane);
            o[n] = wmma_bf16(a, bf, o[n]);
        }
    }
    #pragma unroll
    for (int n = 0; n < 3; ++n)
        #pragma unroll
        for (int r = 0; r < 8; ++r) {
            const int row = mrow + ((lane >> 4) << 3) + r;
            const int col = n * 16 + (lane & 15);
            attnOutB[(size_t)(b * K_ + row) * H_ + h * HD_ + col] =
                (__bf16)o[n][r];
        }
}

// ---------------------------------------------------------------------------
// Residual + LayerNorm: out = LN(x + y) * g + b. One wave per row (H=192).
// Optionally also emits a bf16 copy (input to the next GEMM).
// ---------------------------------------------------------------------------
template<bool OUT_BF16>
__global__ __launch_bounds__(256)
void add_ln_kernel(const float* __restrict__ x, const float* __restrict__ y,
                   const float* __restrict__ g, const float* __restrict__ bta,
                   float* __restrict__ out, __bf16* __restrict__ outB) {
    const int lane = threadIdx.x & 31;
    const int wave = threadIdx.x >> 5;
    const int row  = blockIdx.x * 8 + wave;

    const float* xr = x + (size_t)row * H_;
    const float* yr = y + (size_t)row * H_;

    float v[6];
    float s = 0.0f;
    #pragma unroll
    for (int i = 0; i < 6; ++i) {
        v[i] = xr[lane + 32 * i] + yr[lane + 32 * i];
        s += v[i];
    }
    #pragma unroll
    for (int off = 16; off >= 1; off >>= 1) s += __shfl_xor(s, off, 32);
    const float mean = s * (1.0f / H_);

    float vs = 0.0f;
    #pragma unroll
    for (int i = 0; i < 6; ++i) {
        const float d = v[i] - mean;
        vs += d * d;
    }
    #pragma unroll
    for (int off = 16; off >= 1; off >>= 1) vs += __shfl_xor(vs, off, 32);
    const float rstd = rsqrtf(vs * (1.0f / H_) + LN_EPS);

    float*  orow  = out + (size_t)row * H_;
    __bf16* obrow = OUT_BF16 ? outB + (size_t)row * H_ : nullptr;
    #pragma unroll
    for (int i = 0; i < 6; ++i) {
        const int c = lane + 32 * i;
        const float val = (v[i] - mean) * rstd * g[c] + bta[c];
        orow[c] = val;
        if (OUT_BF16) obrow[c] = (__bf16)val;
    }
}

// ---------------------------------------------------------------------------
// Host launcher
// ---------------------------------------------------------------------------
extern "C" void kernel_launch(void* const* d_in, const int* in_sizes, int n_in,
                              void* d_out, int out_size, void* d_ws, size_t ws_size,
                              hipStream_t stream) {
    (void)in_sizes; (void)n_in; (void)out_size; (void)ws_size;

    const float* h_v    = (const float*)d_in[0];
    const float* cap_v  = (const float*)d_in[1];
    const float* fixedr = (const float*)d_in[2];
    const float* unitr  = (const float*)d_in[3];
    const float* tierr  = (const float*)d_in[4];
    const float* tierraw= (const float*)d_in[5];
    // d_in[6] = veh_batch (unused), d_in[7] = B (unused)
    const float* Wqkv = (const float*)d_in[8];
    const float* Wout = (const float*)d_in[9];
    const float* pW1  = (const float*)d_in[10];
    const float* pb1  = (const float*)d_in[11];
    const float* pW2  = (const float*)d_in[12];
    const float* pb2  = (const float*)d_in[13];
    const float* ln1g = (const float*)d_in[14];
    const float* ln1b = (const float*)d_in[15];
    const float* ln2g = (const float*)d_in[16];
    const float* ln2b = (const float*)d_in[17];
    const float* fW1  = (const float*)d_in[18];
    const float* fb1  = (const float*)d_in[19];
    const float* fW2  = (const float*)d_in[20];
    const float* fb2  = (const float*)d_in[21];

    // ---- workspace layout (byte offsets, 256-aligned, with reuse) ----
    char* w = (char*)d_ws;
    size_t o = 0;
    auto alloc = [&](size_t bytes) -> void* {
        void* p = w + o;
        o = (o + bytes + 255) & ~(size_t)255;
        return p;
    };
    __bf16* h_vb  = (__bf16*)alloc((size_t)N_ * H_ * 2);
    __bf16* qkvB  = (__bf16*)alloc((size_t)N_ * 3 * H_ * 2);          // -> proj
    __bf16* biasB = (__bf16*)alloc((size_t)B_ * NH_ * K_ * K_ * 2);   // -> f2
    __bf16* attnB = (__bf16*)alloc((size_t)N_ * H_ * 2);              // -> hv1b
    float*  hv1   = (float*) alloc((size_t)N_ * H_ * 4);
    __bf16* mid   = (__bf16*)alloc((size_t)N_ * FFN_ * 2);
    __bf16* WqkvT = (__bf16*)alloc((size_t)3 * H_ * H_ * 2);
    __bf16* WoutT = (__bf16*)alloc((size_t)H_ * H_ * 2);
    __bf16* fW1T  = (__bf16*)alloc((size_t)FFN_ * H_ * 2);
    __bf16* fW2T  = (__bf16*)alloc((size_t)H_ * FFN_ * 2);
    float*  proj  = (float*)qkvB;    // qkv dead after attention
    float*  f2    = (float*)biasB;   // bias dead after attention
    __bf16* hv1b  = attnB;           // attn_out dead after proj GEMM

    // ---- one-time conversions ----
    f32_to_bf16_kernel<<<(N_ * H_ / 4 + 255) / 256, 256, 0, stream>>>(
        h_v, h_vb, N_ * H_ / 4);
    transpose_cvt_kernel<<<(H_ * 3 * H_ + 255) / 256, 256, 0, stream>>>(
        Wqkv, WqkvT, H_, 3 * H_);
    transpose_cvt_kernel<<<(H_ * H_ + 255) / 256, 256, 0, stream>>>(
        Wout, WoutT, H_, H_);
    transpose_cvt_kernel<<<(H_ * FFN_ + 255) / 256, 256, 0, stream>>>(
        fW1, fW1T, H_, FFN_);
    transpose_cvt_kernel<<<(FFN_ * H_ + 255) / 256, 256, 0, stream>>>(
        fW2, fW2T, FFN_, H_);

    // 1. pairwise bias MLP (bf16 out)
    pair_bias_kernel<<<B_, 256, 0, stream>>>(cap_v, fixedr, unitr, tierr,
                                             tierraw, pW1, pb1, pW2, pb2,
                                             biasB);
    // 2. qkv = h_v @ Wqkv  (bf16 out)
    gemm_bf16_kernel<false, false, false, true>
        <<<dim3(N_ / 128, (3 * H_) / 64), 128, 0, stream>>>(
        h_vb, WqkvT, nullptr, nullptr, qkvB, H_, 3 * H_);
    // 3. fused attention
    attn_kernel<<<B_ * NH_, 256, 0, stream>>>(qkvB, biasB, attnB);
    // 4. proj = attn_out @ Wout  (f32 out for LN residual)
    gemm_bf16_kernel<false, false, true, false>
        <<<dim3(N_ / 128, H_ / 64), 128, 0, stream>>>(
        attnB, WoutT, nullptr, proj, nullptr, H_, H_);
    // 5. hv1 = LN(h_v + proj)  (f32 + bf16)
    add_ln_kernel<true><<<N_ / 8, 256, 0, stream>>>(
        h_v, proj, ln1g, ln1b, hv1, hv1b);
    // 6. mid = silu(hv1 @ fW1 + fb1)  (bf16 out)
    gemm_bf16_kernel<true, true, false, true>
        <<<dim3(N_ / 128, FFN_ / 64), 128, 0, stream>>>(
        hv1b, fW1T, fb1, nullptr, mid, H_, FFN_);
    // 7. f2 = mid @ fW2 + fb2  (f32 out)
    gemm_bf16_kernel<true, false, true, false>
        <<<dim3(N_ / 128, H_ / 64), 128, 0, stream>>>(
        mid, fW2T, fb2, f2, nullptr, FFN_, H_);
    // 8. out = LN(hv1 + f2)
    add_ln_kernel<false><<<N_ / 8, 256, 0, stream>>>(
        hv1, f2, ln2g, ln2b, (float*)d_out, nullptr);
}
